// QuantumWalkRetriever_86543591014916
// MI455X (gfx1250) — matrix operand order repaired and checked by reading, not compiled, not run on hardware
//
#include <hip/hip_runtime.h>
#include <hip/hip_bf16.h>
#include <math.h>

// ---------------------------------------------------------------------------
// QuantumWalkRetriever for gfx1250 (MI455X)
//   N=100000, D=384, K=8, HID=128, STEPS=3
//
// Algebra: inp = [emb | qv] -> qv half of inp@[W1;P1] is a per-column constant
// (qconst). coin = a a^T / sum(a^2) => coin@s = a * (a.s) * scale.
// Big GEMM: [N,384] @ [384,256] bf16 WMMA, f32 accum, async-LDS staged,
// double-buffered. Walk steps = small bandwidth-trivial kernels.
// ---------------------------------------------------------------------------

#define NN   100000
#define DD   384
#define KK   8
#define HID  128
#define CTOT 256      // HID cols of W1 + HID cols of P1
#define KT   32       // WMMA k-tile (bf16 16x16x32)
#define BSTR 40       // LDS column stride for B (bf16): 80B, 16B-aligned, no bank conflicts
#define ASTR 36       // LDS row stride for A (f32): 144B, 16B-aligned

typedef __attribute__((ext_vector_type(16))) __bf16 v16bf;
typedef __attribute__((ext_vector_type(8)))  __bf16 v8bf;
typedef __attribute__((ext_vector_type(8)))  float  v8f;
typedef __attribute__((ext_vector_type(4)))  float  v4f;
typedef int v4i __attribute__((vector_size(16)));   // matches builtin's V4i

typedef __attribute__((address_space(1))) v4i* gptr_b128;   // global src
typedef __attribute__((address_space(3))) v4i* lptr_b128;   // LDS dst

// ------------------------- ws layout (bytes) -------------------------------
#define WS_QCONST 0
#define WS_WB     1024
#define WS_AMPS   (WS_WB + DD * CTOT * 2)
#define WS_SCALE  (WS_AMPS + NN * KK * 4)
#define WS_STA    (WS_SCALE + NN * 4)
#define WS_STB    (WS_STA + NN * KK * 4)
#define WS_NRM2   (WS_STB + NN * KK * 4)

// --------------------- async global->LDS helpers ---------------------------
#if __has_builtin(__builtin_amdgcn_global_load_async_to_lds_b128)
#define HAS_ASYNC_LDS 1
#else
#define HAS_ASYNC_LDS 0
#endif

__device__ __forceinline__ void async_g2l_b128(const void* g, void* l) {
#if HAS_ASYNC_LDS
  __builtin_amdgcn_global_load_async_to_lds_b128(
      (gptr_b128)g, (lptr_b128)l, 0, 0);
#else
  // synchronous fallback: normal load + ds store (DScnt/loadcnt tracked)
  *(v4f*)l = *(const v4f*)g;
#endif
}

__device__ __forceinline__ void wait_async_6() {
#if HAS_ASYNC_LDS
#if __has_builtin(__builtin_amdgcn_s_wait_asynccnt)
  __builtin_amdgcn_s_wait_asynccnt(6);
#else
  asm volatile("s_wait_asynccnt 0x6" ::: "memory");
#endif
#endif
}

__device__ __forceinline__ void wait_async_0() {
#if HAS_ASYNC_LDS
#if __has_builtin(__builtin_amdgcn_s_wait_asynccnt)
  __builtin_amdgcn_s_wait_asynccnt(0);
#else
  asm volatile("s_wait_asynccnt 0x0" ::: "memory");
#endif
#endif
}

// ---------------------------------------------------------------------------
// Prep: qconst[c] = bias_c + sum_d qv[d]*Wsel[D+d,c];
//       Wb[c*384+k] = bf16(Wsel[k,c])  (column-major: 64B contiguous / col-tile)
// ---------------------------------------------------------------------------
__global__ void qwr_prep(const float* __restrict__ qv,
                         const float* __restrict__ W1,
                         const float* __restrict__ b1,
                         const float* __restrict__ P1,
                         const float* __restrict__ pb1,
                         float* __restrict__ qconst,
                         __bf16* __restrict__ Wb) {
  int c = blockIdx.x * blockDim.x + threadIdx.x;
  if (c >= CTOT) return;
  const int cc = (c < HID) ? c : (c - HID);
  const float* W = (c < HID) ? W1 : P1;
  float acc = (c < HID) ? b1[cc] : pb1[cc];
  for (int d = 0; d < DD; ++d)
    acc += qv[d] * W[(size_t)(DD + d) * HID + cc];
  qconst[c] = acc;
  for (int k = 0; k < DD; ++k)
    Wb[(size_t)c * DD + k] = (__bf16)W[(size_t)k * HID + cc];
}

__global__ void qwr_init(float* __restrict__ stA, float* __restrict__ stB,
                         float* __restrict__ nrm2) {
  int i = blockIdx.x * blockDim.x + threadIdx.x;
  const float initv = 1.0f / sqrtf((float)NN * (float)KK);
  if (i < NN * KK) { stA[i] = initv; stB[i] = 0.0f; }
  if (i < 4) nrm2[i] = 0.0f;
}

// ---------------------------------------------------------------------------
// WMMA GEMM + fused amps epilogue.
// Block = 256 threads (8 waves), tile = 64 rows x 256 cols.
// Wave w: rows (w>>1)*16, cols (w&1)*128 => 8 accumulator tiles (16x16).
// A & B k-tiles double-buffered in LDS, filled with async global->LDS b128.
// Staging region aliases the 64KB hidden buffer (disjoint in time).
// ---------------------------------------------------------------------------
union QwrSmem {
  struct {
    __bf16 B[2][CTOT * BSTR];   // 2 x 20,480 B
    float  A[2][64 * ASTR];     // 2 x  9,216 B
  } stage;
  float hid[64 * CTOT];         // 65,536 B
};

// Stage one B k-tile: 256 cols x 32 k bf16. Thread t owns column t (64B).
__device__ __forceinline__ void stage_B(const __bf16* __restrict__ Wb,
                                        __bf16* Bb, int k0, int tid) {
  const __bf16* src = Wb + (size_t)tid * DD + k0;
  __bf16* dst = Bb + tid * BSTR;
  #pragma unroll
  for (int i = 0; i < 4; ++i) async_g2l_b128(src + i * 8, dst + i * 8);
}

// Stage one A k-tile: 64 rows x 32 k f32 (512 x 16B chunks; 2 per thread).
__device__ __forceinline__ void stage_A(const float* __restrict__ emb,
                                        float* Ab, int blockRow, int k0,
                                        int tid) {
  #pragma unroll
  for (int i = 0; i < 2; ++i) {
    const int c = tid * 2 + i;
    const int r = c >> 3, ch = c & 7;
    int gr = blockRow + r;
    if (gr >= NN) gr = NN - 1;                    // clamp: EXEC stays all-ones
    async_g2l_b128(emb + (size_t)gr * DD + k0 + ch * 4, Ab + r * ASTR + ch * 4);
  }
}

__launch_bounds__(256)
__global__ void qwr_gemm_amps(const float* __restrict__ emb,
                              const __bf16* __restrict__ Wb,
                              const float* __restrict__ qconst,
                              const float* __restrict__ W2,
                              const float* __restrict__ b2,
                              const float* __restrict__ P2,
                              const float* __restrict__ pb2,
                              const float* __restrict__ ps,
                              float* __restrict__ amps_out,
                              float* __restrict__ scale_out) {
  __shared__ QwrSmem smem;

  const int tid  = threadIdx.x;
  const int wave = tid >> 5;
  const int lane = tid & 31;
  const int row  = lane & 15;
  const int half = lane >> 4;
  const int m0   = (wave >> 1) * 16;              // row sub-tile in [0,64)
  const int cb   = (wave & 1) * 128;              // col base in [0,256)
  const int blockRow = blockIdx.x * 64;

  v8f acc[8] = {};

  // prefetch k-tile 0
  stage_B(Wb, smem.stage.B[0], 0, tid);
  stage_A(emb, smem.stage.A[0], blockRow, 0, tid);

  int buf = 0;
  for (int kt = 0; kt < DD / KT; ++kt) {
    const int k0 = kt * KT;
    if (kt + 1 < DD / KT) {                       // prefetch next tile
      stage_B(Wb, smem.stage.B[buf ^ 1], k0 + KT, tid);
      stage_A(emb, smem.stage.A[buf ^ 1], blockRow, k0 + KT, tid);
      wait_async_6();                             // oldest 6 (tile kt) done
    } else {
      wait_async_0();
    }
    __syncthreads();                              // tile kt visible to all

    // A fragment from LDS, cvt f32->bf16 (lanes 0-15: K = kc..kc+7,kc+16..23)
    const int kc0 = half * 8;
    const float* ar = smem.stage.A[buf] + (m0 + row) * ASTR;
    v4f a0 = *(const v4f*)(ar + kc0);
    v4f a1 = *(const v4f*)(ar + kc0 + 4);
    v4f a2 = *(const v4f*)(ar + kc0 + 16);
    v4f a3 = *(const v4f*)(ar + kc0 + 20);
    v16bf afrag;
    #pragma unroll
    for (int t = 0; t < 4; ++t) {
      afrag[t]      = (__bf16)a0[t];
      afrag[4 + t]  = (__bf16)a1[t];
      afrag[8 + t]  = (__bf16)a2[t];
      afrag[12 + t] = (__bf16)a3[t];
    }

    #pragma unroll
    for (int j = 0; j < 8; ++j) {
      const __bf16* bp = smem.stage.B[buf] + (cb + 16 * j + row) * BSTR + half * 8;
      const v8bf blo = *(const v8bf*)bp;
      const v8bf bhi = *(const v8bf*)(bp + 16);
      const v16bf bfrag = __builtin_shufflevector(
          blo, bhi, 0, 1, 2, 3, 4, 5, 6, 7, 8, 9, 10, 11, 12, 13, 14, 15);
      acc[j] = __builtin_amdgcn_wmma_f32_16x16x32_bf16(
          false, afrag, false, bfrag, (short)0, acc[j], false, false);
    }
    __syncthreads();                              // done reading buf before refill
    buf ^= 1;
  }

  // hidden = relu(acc + qconst) -> LDS (C layout: vgpr r, lane l ->
  // M = half*8 + r, N = l&15). Stage region is dead now (barrier above).
  #pragma unroll
  for (int j = 0; j < 8; ++j) {
    const int col = cb + 16 * j + row;
    const float qc = qconst[col];
    #pragma unroll
    for (int r = 0; r < 8; ++r) {
      const int mm = m0 + half * 8 + r;
      float h = acc[j][r] + qc;
      smem.hid[mm * CTOT + col] = (h > 0.0f) ? h : 0.0f;
    }
  }
  __syncthreads();

  // Per-row epilogue: amps = hid[:,0:128]@W2 + b2 ; plog = hid[:,128:]@P2 + pb2
  if (tid < 64) {
    const int grow = blockRow + tid;
    if (grow < NN) {
      const float* hrow = smem.hid + tid * CTOT;
      float a[KK];
      #pragma unroll
      for (int kk = 0; kk < KK; ++kk) a[kk] = b2[kk];
      float plog = pb2[0];
      for (int c = 0; c < HID; ++c) {
        const float h = hrow[c];
        #pragma unroll
        for (int kk = 0; kk < KK; ++kk) a[kk] += h * W2[c * KK + kk];
        plog += hrow[HID + c] * P2[c];
      }
      const float psv = ps[grow];
      if (psv > 0.0f) {
        const float pf = psv / (1.0f + expf(-plog));    // sigmoid(plog)*ps
        #pragma unroll
        for (int kk = 0; kk < KK; ++kk) a[kk] *= (1.0f + pf);
      }
      float fro = 0.0f;                                  // ||outer||_F = sum a^2
      #pragma unroll
      for (int kk = 0; kk < KK; ++kk) fro += a[kk] * a[kk];
      float sc;
      if (__builtin_isfinite(fro) && fro > 0.0f) {
        sc = 1.0f / fro;
      } else {                                           // uniform coin fallback
        sc = 1.0f / (float)KK;
        #pragma unroll
        for (int kk = 0; kk < KK; ++kk) a[kk] = 1.0f;
      }
      #pragma unroll
      for (int kk = 0; kk < KK; ++kk) amps_out[grow * KK + kk] = a[kk];
      scale_out[grow] = sc;
    }
  }
}

// ---------------------------------------------------------------------------
// Walk step: s_p = a * (a.s) * scale ; scatter-add into next state.
// ---------------------------------------------------------------------------
__global__ void qwr_scatter(const float* __restrict__ cur,
                            float* __restrict__ nxt,
                            const float* __restrict__ amps,
                            const float* __restrict__ scale,
                            const int* __restrict__ nbrs) {
  int i = blockIdx.x * blockDim.x + threadIdx.x;
  if (i >= NN) return;
  float a[KK], s[KK];
  #pragma unroll
  for (int j = 0; j < KK; ++j) { a[j] = amps[i * KK + j]; s[j] = cur[i * KK + j]; }
  float dot = 0.0f;
  #pragma unroll
  for (int j = 0; j < KK; ++j) dot += a[j] * s[j];
  const float f = dot * scale[i];
  #pragma unroll
  for (int j = 0; j < KK; ++j) {
    const int tgt = nbrs[i * KK + j];
    atomicAdd(&nxt[tgt * KK + j], a[j] * f);
  }
}

__global__ void qwr_reduce(const float* __restrict__ v, float* __restrict__ slot) {
  __shared__ float sdata[256];
  float sum = 0.0f;
  for (int i = blockIdx.x * blockDim.x + threadIdx.x; i < NN * KK;
       i += gridDim.x * blockDim.x) {
    const float x = v[i];
    sum += x * x;
  }
  sdata[threadIdx.x] = sum;
  __syncthreads();
  for (int s = 128; s > 0; s >>= 1) {
    if (threadIdx.x < s) sdata[threadIdx.x] += sdata[threadIdx.x + s];
    __syncthreads();
  }
  if (threadIdx.x == 0) atomicAdd(slot, sdata[0]);
}

__global__ void qwr_norm(float* __restrict__ nxt, float* __restrict__ other,
                         const float* __restrict__ slot) {
  int i = blockIdx.x * blockDim.x + threadIdx.x;
  if (i >= NN * KK) return;
  const float n2 = *slot;
  const float initv = 1.0f / sqrtf((float)NN * (float)KK);
  nxt[i] = (n2 > 0.0f) ? nxt[i] / sqrtf(n2) : initv;
  other[i] = 0.0f;
}

__global__ void qwr_final(const float* __restrict__ st, float* __restrict__ out) {
  int i = blockIdx.x * blockDim.x + threadIdx.x;
  if (i >= NN) return;
  float s = 0.0f;
  #pragma unroll
  for (int j = 0; j < KK; ++j) s += fabsf(st[i * KK + j]);
  out[i] = s;
}

// ---------------------------------------------------------------------------
extern "C" void kernel_launch(void* const* d_in, const int* in_sizes, int n_in,
                              void* d_out, int out_size, void* d_ws, size_t ws_size,
                              hipStream_t stream) {
  const float* emb = (const float*)d_in[0];
  const float* qv  = (const float*)d_in[1];
  const int*   nbrs= (const int*)  d_in[2];
  const float* ps  = (const float*)d_in[3];
  const float* W1  = (const float*)d_in[4];
  const float* b1  = (const float*)d_in[5];
  const float* W2  = (const float*)d_in[6];
  const float* b2  = (const float*)d_in[7];
  const float* P1  = (const float*)d_in[8];
  const float* pb1 = (const float*)d_in[9];
  const float* P2  = (const float*)d_in[10];
  const float* pb2 = (const float*)d_in[11];

  char* ws = (char*)d_ws;
  float*  qconst = (float*)(ws + WS_QCONST);
  __bf16* Wb     = (__bf16*)(ws + WS_WB);
  float*  amps   = (float*)(ws + WS_AMPS);
  float*  scale  = (float*)(ws + WS_SCALE);
  float*  stA    = (float*)(ws + WS_STA);
  float*  stB    = (float*)(ws + WS_STB);
  float*  nrm2   = (float*)(ws + WS_NRM2);

  qwr_prep<<<1, 256, 0, stream>>>(qv, W1, b1, P1, pb1, qconst, Wb);
  qwr_init<<<(NN * KK + 255) / 256, 256, 0, stream>>>(stA, stB, nrm2);
  qwr_gemm_amps<<<(NN + 63) / 64, 256, 0, stream>>>(
      emb, Wb, qconst, W2, b2, P2, pb2, ps, amps, scale);

  float* cur = stA;
  float* nxt = stB;
  for (int s = 0; s < 3; ++s) {
    qwr_scatter<<<(NN + 255) / 256, 256, 0, stream>>>(cur, nxt, amps, scale, nbrs);
    qwr_reduce<<<512, 256, 0, stream>>>(nxt, nrm2 + s);
    qwr_norm<<<(NN * KK + 255) / 256, 256, 0, stream>>>(nxt, cur, nrm2 + s);
    float* t = cur; cur = nxt; nxt = t;
  }
  qwr_final<<<(NN + 255) / 256, 256, 0, stream>>>(cur, (float*)d_out);
}
